// WSAD_42288247996461
// MI455X (gfx1250) — matrix operand order repaired and checked by reading, not compiled, not run on hardware
//
#include <hip/hip_runtime.h>
#include <hip/hip_bf16.h>
#include <cstdint>

#define DEV __device__ __forceinline__

typedef __bf16 bf16_t;
typedef __attribute__((ext_vector_type(16))) bf16_t v16bf;
typedef __attribute__((ext_vector_type(2)))  bf16_t v2bf;
typedef __attribute__((ext_vector_type(8)))  float  v8f;

// ---------------- problem constants ----------------
constexpr int B_NOR = 16, B_ABN = 16, NCR = 5, T_ = 256, D_IN = 1024, DE = 512;
constexpr int MEM_N = 60, MEM_NP = 64, KSEL = 17;           // KSEL = T/16+1
constexpr int BN   = (B_NOR + B_ABN) * NCR;                  // 160
constexpr int NBH  = B_NOR * NCR;                            // 80 rows per half
constexpr int MALL = BN * T_;                                // 40960
constexpr int MHALF = NBH * T_;                              // 20480
constexpr int DCAT = 2 * DE, CAR = 64, CLSH = 128;
constexpr int KCONV = 3 * D_IN;                              // 3072

// ---------------- helpers ----------------
DEV unsigned short f2bf_u(float f) {
  union { float f; unsigned u; } a; a.f = f;
  unsigned r = a.u + 0x7FFFu + ((a.u >> 16) & 1u);
  return (unsigned short)(r >> 16);
}
DEV unsigned pack_bf2(float lo, float hi) {
#if __has_builtin(__builtin_amdgcn_cvt_pk_bf16_f32)
  union { v2bf v; unsigned u; } o;
  o.v = __builtin_amdgcn_cvt_pk_bf16_f32(lo, hi);
  return o.u;
#else
  return ((unsigned)f2bf_u(hi) << 16) | (unsigned)f2bf_u(lo);
#endif
}
DEV void st_bf(bf16_t* p, float v) { *(unsigned short*)p = f2bf_u(v); }
DEV float bf2f(const bf16_t* p) {
  union { unsigned u; float f; } o; o.u = ((unsigned)(*(const unsigned short*)p)) << 16; return o.f;
}
DEV float sigmoidf_(float x) { return 1.f / (1.f + expf(-x)); }

// A fragment (16x32 bf16, MxK) from row-major [M][K]; two 16B contiguous runs per lane
DEV v16bf ld_a(const bf16_t* A, int lda, int row0, int k0, int lane) {
  union { v16bf v; uint4 q[2]; } f;
  int m = lane & 15;
  int kb = (lane < 16) ? 0 : 8;
  const bf16_t* p = A + (size_t)(row0 + m) * lda + k0 + kb;
  f.q[0] = *(const uint4*)(p);
  f.q[1] = *(const uint4*)(p + 16);
  return f.v;
}
// B fragment (32x16 bf16, KxN) from K-contiguous transposed weights Bt[N][K]; two 16B contiguous runs per lane
DEV v16bf ld_b(const bf16_t* Bt, int ldb, int col0, int k0, int lane) {
  union { v16bf v; uint4 q[2]; } f;
  int n = lane & 15;
  int kb = (lane < 16) ? 0 : 8;
  const bf16_t* p = Bt + (size_t)(col0 + n) * ldb + k0 + kb;
  f.q[0] = *(const uint4*)(p);
  f.q[1] = *(const uint4*)(p + 16);
  return f.v;
}

DEV float blk_sum(float v, float* red) {
  int t = threadIdx.x;
  red[t] = v; __syncthreads();
  for (int s = blockDim.x >> 1; s > 0; s >>= 1) { if (t < s) red[t] += red[t + s]; __syncthreads(); }
  float r = red[0]; __syncthreads();
  return r;
}
DEV float blk_max(float v, float* red) {
  int t = threadIdx.x;
  red[t] = v; __syncthreads();
  for (int s = blockDim.x >> 1; s > 0; s >>= 1) { if (t < s) red[t] = fmaxf(red[t], red[t + s]); __syncthreads(); }
  float r = red[0]; __syncthreads();
  return r;
}

// ---------------- prep kernels ----------------
__global__ void k_prep_conv(const float* __restrict__ cw, bf16_t* __restrict__ wt) {
  int tid = blockIdx.x * blockDim.x + threadIdx.x;
  if (tid >= DE * KCONV) return;
  int e = tid / KCONV, kk = tid - e * KCONV;
  int kg = kk >> 10, c = kk & 1023;                 // kk = kg*1024 + c
  st_bf(wt + tid, cw[(size_t)e * D_IN * 3 + (size_t)c * 3 + kg]);
}
__global__ void k_prep_mem(const float* __restrict__ am, const float* __restrict__ nm,
                           bf16_t* amb, bf16_t* nmb, bf16_t* amT, bf16_t* nmT) {
  int tid = blockIdx.x * blockDim.x + threadIdx.x;
  if (tid >= MEM_NP * DE) return;
  int n = tid / DE, d = tid - n * DE;
  float va = (n < MEM_N) ? am[(size_t)n * DE + d] : 0.f;
  float vn = (n < MEM_N) ? nm[(size_t)n * DE + d] : 0.f;
  st_bf(amb + (size_t)n * DE + d, va);
  st_bf(nmb + (size_t)n * DE + d, vn);
  st_bf(amT + (size_t)d * MEM_NP + n, va);
  st_bf(nmT + (size_t)d * MEM_NP + n, vn);
}
__global__ void k_prep_t512(const float* __restrict__ mw, const float* __restrict__ vw,
                            bf16_t* muT, bf16_t* varT) {
  int tid = blockIdx.x * blockDim.x + threadIdx.x;
  if (tid >= DE * DE) return;
  int k = tid / DE, n = tid - k * DE;
  st_bf(muT  + (size_t)n * DE + k, mw[tid]);
  st_bf(varT + (size_t)n * DE + k, vw[tid]);
}
__global__ void k_prep_w1(const float* __restrict__ w1, bf16_t* w1T) {
  int tid = blockIdx.x * blockDim.x + threadIdx.x;
  if (tid >= DCAT * CLSH) return;
  int k = tid / CLSH, h = tid - k * CLSH;
  st_bf(w1T + (size_t)h * DCAT + k, w1[tid]);
}
__global__ void k_zero(float* p, int n) {
  int i = blockIdx.x * blockDim.x + threadIdx.x;
  if (i < n) p[i] = 0.f;
}

// ---------------- conv GEMM (im2col, fused ReLU, f32 + bf16 outputs) ----------------
__global__ void k_conv_gemm(const float* __restrict__ x, const bf16_t* __restrict__ wt,
                            const float* __restrict__ bias, float* __restrict__ xe,
                            bf16_t* __restrict__ xeb) {
  int wid = blockIdx.x * (blockDim.x >> 5) + (threadIdx.x >> 5);
  int lane = threadIdx.x & 31;
  constexpr int NT = DE / 16;
  int tm = wid / NT, tn = wid - tm * NT;
  if (tm >= MALL / 16) return;
  int row0 = tm * 16, col0 = tn * 16;
  v8f acc = {};
  int m = lane & 15;
  int r = row0 + m;
  int bn = r >> 8, t = r & 255;
  int kb = (lane < 16) ? 0 : 8;
  int nB = lane & 15;
  for (int k0 = 0; k0 < KCONV; k0 += 32) {
    union { v16bf v; unsigned u[8]; } fa;
#pragma unroll
    for (int h = 0; h < 2; ++h) {
      int kk = k0 + kb + h * 16;           // 8 contiguous K elements, same tap group
      int kg = kk >> 10, c = kk & 1023;
      int tt = t + kg - 1;
      bool ok = ((unsigned)tt < (unsigned)T_);
      int tc = ok ? tt : 0;
      const float* p = x + ((size_t)bn * T_ + tc) * D_IN + c;
      float4 f0 = *(const float4*)(p);
      float4 f1 = *(const float4*)(p + 4);
      float mk = ok ? 1.f : 0.f;
      fa.u[4 * h + 0] = pack_bf2(f0.x * mk, f0.y * mk);
      fa.u[4 * h + 1] = pack_bf2(f0.z * mk, f0.w * mk);
      fa.u[4 * h + 2] = pack_bf2(f1.x * mk, f1.y * mk);
      fa.u[4 * h + 3] = pack_bf2(f1.z * mk, f1.w * mk);
    }
    v16bf fb = ld_b(wt, KCONV, col0, k0, lane);
    acc = __builtin_amdgcn_wmma_f32_16x16x32_bf16(false, fa.v, false, fb, (short)0, acc, false, false);
  }
  int mo = (lane < 16) ? 0 : 8;
  float bv = bias[col0 + nB];
#pragma unroll
  for (int v = 0; v < 8; ++v) {
    size_t off = (size_t)(row0 + mo + v) * DE + col0 + nB;
    float val = acc[v] + bv; if (val < 0.f) val = 0.f;
    xe[off] = val; st_bf(xeb + off, val);
  }
}

// ---------------- attention GEMM: sigmoid(z @ mem^T / sqrt(512)) ----------------
__global__ void k_att_gemm(const bf16_t* __restrict__ A, const bf16_t* __restrict__ memb,
                           bf16_t* __restrict__ att) {
  int wid = blockIdx.x * (blockDim.x >> 5) + (threadIdx.x >> 5);
  int lane = threadIdx.x & 31;
  constexpr int NT = MEM_NP / 16;
  int tm = wid / NT, tn = wid - tm * NT;
  if (tm >= MHALF / 16) return;
  int row0 = tm * 16, col0 = tn * 16;
  v8f acc = {};
  for (int k0 = 0; k0 < DE; k0 += 32) {
    v16bf fa = ld_a(A, DE, row0, k0, lane);
    v16bf fb = ld_b(memb, DE, col0, k0, lane);
    acc = __builtin_amdgcn_wmma_f32_16x16x32_bf16(false, fa, false, fb, (short)0, acc, false, false);
  }
  const float invs = 0.044194173824159216f;  // 1/sqrt(512)
  int n = lane & 15, mo = (lane < 16) ? 0 : 8;
#pragma unroll
  for (int v = 0; v < 8; ++v) {
    float val = (col0 + n < MEM_N) ? sigmoidf_(acc[v] * invs) : 0.f;
    st_bf(att + (size_t)(row0 + mo + v) * MEM_NP + col0 + n, val);
  }
}

// ---------------- t_att = mean(top4 of 60) ----------------
__global__ void k_tatt(const bf16_t* __restrict__ att, float* __restrict__ tatt) {
  int r = blockIdx.x * blockDim.x + threadIdx.x;
  if (r >= MHALF) return;
  float t0 = -1e30f, t1 = -1e30f, t2 = -1e30f, t3 = -1e30f;
  const bf16_t* row = att + (size_t)r * MEM_NP;
  for (int j = 0; j < MEM_N; ++j) {
    float v = bf2f(row + j);
    if (v > t0) { t3 = t2; t2 = t1; t1 = t0; t0 = v; }
    else if (v > t1) { t3 = t2; t2 = t1; t1 = v; }
    else if (v > t2) { t3 = t2; t2 = v; }
    else if (v > t3) { t3 = v; }
  }
  tatt[r] = (t0 + t1 + t2 + t3) * 0.25f;
}

// ---------------- aug GEMM: att @ mem ----------------
__global__ void k_aug_gemm(const bf16_t* __restrict__ att, const bf16_t* __restrict__ memT,
                           bf16_t* __restrict__ aug) {
  int wid = blockIdx.x * (blockDim.x >> 5) + (threadIdx.x >> 5);
  int lane = threadIdx.x & 31;
  constexpr int NT = DE / 16;
  int tm = wid / NT, tn = wid - tm * NT;
  if (tm >= MHALF / 16) return;
  int row0 = tm * 16, col0 = tn * 16;
  v8f acc = {};
  for (int k0 = 0; k0 < MEM_NP; k0 += 32) {
    v16bf fa = ld_a(att, MEM_NP, row0, k0, lane);
    v16bf fb = ld_b(memT, MEM_NP, col0, k0, lane);
    acc = __builtin_amdgcn_wmma_f32_16x16x32_bf16(false, fa, false, fb, (short)0, acc, false, false);
  }
  int n = lane & 15, mo = (lane < 16) ? 0 : 8;
#pragma unroll
  for (int v = 0; v < 8; ++v)
    st_bf(aug + (size_t)(row0 + mo + v) * DE + col0 + n, acc[v]);
}

// ---------------- mu/var GEMMs fused with reparameterization + KL ----------------
__global__ void k_muvar(const bf16_t* __restrict__ aug, const bf16_t* __restrict__ muT,
                        const bf16_t* __restrict__ varT, const float* __restrict__ mub,
                        const float* __restrict__ varb, const float* __restrict__ eps,
                        float* __restrict__ nnew, float* __restrict__ klacc) {
  int wid = blockIdx.x * (blockDim.x >> 5) + (threadIdx.x >> 5);
  int lane = threadIdx.x & 31;
  constexpr int NT = DE / 16;
  int tm = wid / NT, tn = wid - tm * NT;
  if (tm >= MHALF / 16) return;
  int row0 = tm * 16, col0 = tn * 16;
  v8f a1 = {}, a2 = {};
  for (int k0 = 0; k0 < DE; k0 += 32) {
    v16bf fa = ld_a(aug, DE, row0, k0, lane);
    v16bf b1 = ld_b(muT, DE, col0, k0, lane);
    v16bf b2 = ld_b(varT, DE, col0, k0, lane);
    a1 = __builtin_amdgcn_wmma_f32_16x16x32_bf16(false, fa, false, b1, (short)0, a1, false, false);
    a2 = __builtin_amdgcn_wmma_f32_16x16x32_bf16(false, fa, false, b2, (short)0, a2, false, false);
  }
  int n = lane & 15, mo = (lane < 16) ? 0 : 8;
  float mb = mub[col0 + n], vb = varb[col0 + n];
  float kl = 0.f;
#pragma unroll
  for (int v = 0; v < 8; ++v) {
    size_t off = (size_t)(row0 + mo + v) * DE + col0 + n;
    float mu = a1[v] + mb, var = a2[v] + vb;
    float ev = expf(var);
    nnew[off] = mu + eps[off] * sqrtf(ev);
    kl += 1.f + var - mu * mu - ev;
  }
  for (int s = 16; s >= 1; s >>= 1) kl += __shfl_xor(kl, s, 32);
  if (lane == 0) atomicAdd(klacc, kl);
}

// ---------------- aug @ mu_w + mu_b (+ optional base) ----------------
__global__ void k_gemm512(const bf16_t* __restrict__ aug, const bf16_t* __restrict__ muT,
                          const float* __restrict__ mub, const float* __restrict__ base,
                          float* __restrict__ out) {
  int wid = blockIdx.x * (blockDim.x >> 5) + (threadIdx.x >> 5);
  int lane = threadIdx.x & 31;
  constexpr int NT = DE / 16;
  int tm = wid / NT, tn = wid - tm * NT;
  if (tm >= MHALF / 16) return;
  int row0 = tm * 16, col0 = tn * 16;
  v8f acc = {};
  for (int k0 = 0; k0 < DE; k0 += 32) {
    v16bf fa = ld_a(aug, DE, row0, k0, lane);
    v16bf fb = ld_b(muT, DE, col0, k0, lane);
    acc = __builtin_amdgcn_wmma_f32_16x16x32_bf16(false, fa, false, fb, (short)0, acc, false, false);
  }
  int n = lane & 15, mo = (lane < 16) ? 0 : 8;
  float mb = mub[col0 + n];
#pragma unroll
  for (int v = 0; v < 8; ++v) {
    size_t off = (size_t)(row0 + mo + v) * DE + col0 + n;
    float val = acc[v] + mb;
    if (base) val += base[off];
    out[off] = val;
  }
}

// ---------------- top-17 indices (descending, earliest on ties) ----------------
__global__ void k_topk(const float* __restrict__ tatt, int* __restrict__ idx) {
  int r = blockIdx.x * blockDim.x + threadIdx.x;
  if (r >= NBH) return;
  const float* row = tatt + (size_t)r * T_;
  unsigned long long taken[4] = {0ull, 0ull, 0ull, 0ull};
  for (int s = 0; s < KSEL; ++s) {
    float best = -3.4e38f; int bi = 0;
    for (int t = 0; t < T_; ++t) {
      if ((taken[t >> 6] >> (t & 63)) & 1ull) continue;
      float v = row[t];
      if (v > best) { best = v; bi = t; }
    }
    taken[bi >> 6] |= 1ull << (bi & 63);
    idx[r * KSEL + s] = bi;
  }
}

// ---------------- gather + crop/topk mean -> (16,512) ----------------
__global__ void k_gathermean(const float* __restrict__ feat, const int* __restrict__ idx,
                             float* __restrict__ out) {
  int tid = blockIdx.x * blockDim.x + threadIdx.x;
  if (tid >= 16 * DE) return;
  int b2 = tid >> 9, d = tid & (DE - 1);
  float s = 0.f;
  for (int n = 0; n < NCR; ++n) {
    int r = b2 * NCR + n;
    for (int k = 0; k < KSEL; ++k)
      s += feat[((size_t)r * T_ + idx[r * KSEL + k]) * DE + d];
  }
  out[tid] = s * (1.f / (NCR * KSEL));
}

// ---------------- triplet on l2-normalized (16,512) vectors ----------------
__global__ void k_triplet(const float* __restrict__ anc, const float* __restrict__ pos,
                          const float* __restrict__ neg, float* __restrict__ out) {
  __shared__ float red[256];
  float loss = 0.f;
  for (int b = 0; b < 16; ++b) {
    const float* a = anc + (size_t)b * DE;
    const float* p = pos + (size_t)b * DE;
    const float* n = neg + (size_t)b * DE;
    float sa = 0, sp = 0, sn = 0;
    for (int d = threadIdx.x; d < DE; d += blockDim.x) {
      sa += a[d] * a[d]; sp += p[d] * p[d]; sn += n[d] * n[d];
    }
    sa = blk_sum(sa, red); sp = blk_sum(sp, red); sn = blk_sum(sn, red);
    float ia = 1.f / sqrtf(sa), ip = 1.f / sqrtf(sp), in_ = 1.f / sqrtf(sn);
    float dap = 0, dan = 0;
    for (int d = threadIdx.x; d < DE; d += blockDim.x) {
      float av = a[d] * ia;
      float dv = av - p[d] * ip + 1e-6f; dap += dv * dv;
      float ev = av - n[d] * in_ + 1e-6f; dan += ev * ev;
    }
    dap = blk_sum(dap, red); dan = blk_sum(dan, red);
    float l = sqrtf(dap) - sqrtf(dan) + 1.f;
    loss += (l > 0.f ? l : 0.f);
  }
  if (threadIdx.x == 0) out[0] = loss * (1.f / 16.f);
}

// ---------------- per-(b2,k) gathered-mean magnitudes ----------------
__global__ void k_mag(const float* __restrict__ buf, const int* __restrict__ idx,
                      float* __restrict__ mag) {
  int w = blockIdx.x;
  int b2 = w / KSEL, k = w - b2 * KSEL;
  int lane = threadIdx.x;
  float s = 0.f;
  for (int d = lane; d < DE; d += 32) {
    float m = 0.f;
    for (int n = 0; n < NCR; ++n) {
      int r = b2 * NCR + n;
      m += buf[((size_t)r * T_ + idx[r * KSEL + k]) * DE + d];
    }
    m *= (1.f / NCR);
    s += m * m;
  }
  for (int sh = 16; sh >= 1; sh >>= 1) s += __shfl_xor(s, sh, 32);
  if (lane == 0) mag[w] = sqrtf(s);
}

__global__ void k_distance(const float* __restrict__ nor, const float* __restrict__ abn,
                           float* __restrict__ out) {
  __shared__ float red[256];
  float sp = 0.f, sn = 0.f;
  for (int i = threadIdx.x; i < 16 * KSEL; i += blockDim.x) {
    float a = 1e-6f - nor[i]; sp += a * a;
    float b = 1e-6f - abn[i]; sn += b * b;
  }
  sp = blk_sum(sp, red); sn = blk_sum(sn, red);
  if (threadIdx.x == 0) {
    float l = sqrtf(sp) - sqrtf(sn) + 15.f;
    out[0] = l > 0.f ? l : 0.f;
  }
}

__global__ void k_klfin(const float* __restrict__ acc, float* __restrict__ out) {
  out[0] = -0.5f * acc[0] / (80.f * 512.f);
}

// ---------------- channel attention ----------------
__global__ void k_pool(const float* __restrict__ xe, const float* __restrict__ xh,
                       float* __restrict__ pool) {
  int tid = blockIdx.x * blockDim.x + threadIdx.x;
  if (tid >= BN * DCAT) return;
  int bn = tid >> 10, d = tid & 1023;
  float s = 0.f;
  if (d < DE) { for (int t = 0; t < T_; ++t) s += xe[((size_t)bn * T_ + t) * DE + d]; }
  else { int dd = d - DE; for (int t = 0; t < T_; ++t) s += xh[((size_t)bn * T_ + t) * DE + dd]; }
  pool[tid] = s * (1.f / T_);
}
__global__ void k_cahid(const float* __restrict__ pool, const float* __restrict__ w1,
                        const float* __restrict__ b1, float* __restrict__ hid) {
  int tid = blockIdx.x * blockDim.x + threadIdx.x;
  if (tid >= BN * CAR) return;
  int bn = tid / CAR, h = tid - bn * CAR;
  float s = b1[h];
  for (int k = 0; k < DCAT; ++k) s += pool[(size_t)bn * DCAT + k] * w1[(size_t)k * CAR + h];
  hid[tid] = s > 0.f ? s : 0.f;
}
__global__ void k_catt(const float* __restrict__ hid, const float* __restrict__ w2,
                       const float* __restrict__ b2, float* __restrict__ catt) {
  int tid = blockIdx.x * blockDim.x + threadIdx.x;
  if (tid >= BN * DCAT) return;
  int bn = tid >> 10, d = tid & 1023;
  float s = b2[d];
  for (int h = 0; h < CAR; ++h) s += hid[(size_t)bn * CAR + h] * w2[(size_t)h * DCAT + d];
  catt[tid] = sigmoidf_(s);
}

// ---------------- classifier: gated xc -> 128 hidden (WMMA) -> sigmoid score ----------------
__global__ void k_cls(const float* __restrict__ xe, const float* __restrict__ xh,
                      const float* __restrict__ catt, const bf16_t* __restrict__ w1T,
                      const float* __restrict__ b1, const float* __restrict__ w2,
                      const float* __restrict__ b2, float* __restrict__ score, int inv) {
  int wid = blockIdx.x * (blockDim.x >> 5) + (threadIdx.x >> 5);
  int lane = threadIdx.x & 31;
  if (wid >= MALL / 16) return;
  int row0 = wid * 16;
  v8f z = {};
  v8f acc[8];
#pragma unroll
  for (int i = 0; i < 8; ++i) acc[i] = z;
  int m = lane & 15;
  int r = row0 + m;
  int bn = r >> 8;
  int kb = (lane < 16) ? 0 : 8;
  for (int k0 = 0; k0 < DCAT; k0 += 32) {
    union { v16bf v; unsigned u[8]; } fa;
#pragma unroll
    for (int h = 0; h < 2; ++h) {
      int k = k0 + kb + h * 16;            // 8 contiguous, never straddles the 512 concat split
      const float* src = (k < DE) ? (xe + (size_t)r * DE + k) : (xh + (size_t)r * DE + (k - DE));
      const float* cs = catt + (size_t)bn * DCAT + k;
      float4 a0 = *(const float4*)(src);
      float4 a1 = *(const float4*)(src + 4);
      float4 c0 = *(const float4*)(cs);
      float4 c1 = *(const float4*)(cs + 4);
      if (inv) {
        c0.x = 1.f - c0.x; c0.y = 1.f - c0.y; c0.z = 1.f - c0.z; c0.w = 1.f - c0.w;
        c1.x = 1.f - c1.x; c1.y = 1.f - c1.y; c1.z = 1.f - c1.z; c1.w = 1.f - c1.w;
      }
      fa.u[4 * h + 0] = pack_bf2(a0.x * c0.x, a0.y * c0.y);
      fa.u[4 * h + 1] = pack_bf2(a0.z * c0.z, a0.w * c0.w);
      fa.u[4 * h + 2] = pack_bf2(a1.x * c1.x, a1.y * c1.y);
      fa.u[4 * h + 3] = pack_bf2(a1.z * c1.z, a1.w * c1.w);
    }
#pragma unroll
    for (int nt = 0; nt < 8; ++nt) {
      v16bf fb = ld_b(w1T, DCAT, nt * 16, k0, lane);
      acc[nt] = __builtin_amdgcn_wmma_f32_16x16x32_bf16(false, fa.v, false, fb, (short)0, acc[nt], false, false);
    }
  }
  int n = lane & 15, mo = (lane < 16) ? 0 : 8;
  float sc8[8];
#pragma unroll
  for (int v = 0; v < 8; ++v) sc8[v] = 0.f;
#pragma unroll
  for (int nt = 0; nt < 8; ++nt) {
    int h = nt * 16 + n;
    float bb = b1[h], ww = w2[h];
#pragma unroll
    for (int v = 0; v < 8; ++v) {
      float hid = acc[nt][v] + bb;
      if (hid < 0.f) hid = 0.f;
      sc8[v] += hid * ww;
    }
  }
#pragma unroll
  for (int v = 0; v < 8; ++v) {
    float s = sc8[v];
    s += __shfl_xor(s, 1, 32); s += __shfl_xor(s, 2, 32);
    s += __shfl_xor(s, 4, 32); s += __shfl_xor(s, 8, 32);
    if (n == 0) score[row0 + mo + v] = sigmoidf_(s + b2[0]);
  }
}

// ---------------- attention-score crop means -> d_out (16,256) ----------------
__global__ void k_attmean(const float* __restrict__ tatt, float* __restrict__ out) {
  int tid = blockIdx.x * blockDim.x + threadIdx.x;
  if (tid >= 16 * T_) return;
  int b2 = tid >> 8, t = tid & 255;
  float s = 0.f;
  for (int n = 0; n < NCR; ++n) s += tatt[(size_t)(b2 * NCR + n) * T_ + t];
  out[tid] = s * 0.2f;
}

// ---------------- final: crop means, softmax bags, frame ----------------
__global__ void k_final(const float* __restrict__ sce, const float* __restrict__ scs,
                        const float* __restrict__ t_na, const float* __restrict__ t_aa,
                        float* __restrict__ frame, float* __restrict__ bee,
                        float* __restrict__ bes, float* __restrict__ bse,
                        float* __restrict__ bss) {
  __shared__ float red[256];
  int b = blockIdx.x, t = threadIdx.x;
  float se = 0.f, ss = 0.f, te = 0.f;
  for (int n = 0; n < NCR; ++n) {
    int r = b * NCR + n;
    se += sce[(size_t)r * T_ + t];
    ss += scs[(size_t)r * T_ + t];
    te += (b < 16) ? t_na[(size_t)r * T_ + t]
                   : t_aa[(size_t)((b - 16) * NCR + n) * T_ + t];
  }
  se *= 0.2f; ss *= 0.2f; te *= 0.2f;
  float ts = 1.f - te;
  float me = blk_max(te, red); float ee = expf(te - me);
  float sume = blk_sum(ee, red); float sme = ee / sume;
  float ms = blk_max(ts, red); float es = expf(ts - ms);
  float sums = blk_sum(es, red); float sms = es / sums;
  float v_ee = blk_sum(se * sme, red);
  float v_es = blk_sum(se * sms, red);
  float v_se = blk_sum(ss * sme, red);
  float v_ss = blk_sum(ss * sms, red);
  frame[(size_t)b * T_ + t] = se * te;
  if (t == 0) { bee[b] = v_ee; bes[b] = v_es; bse[b] = v_se; bss[b] = v_ss; }
}

// ---------------- workspace layout (~395 MB) ----------------
constexpr size_t AL(size_t x) { return (x + 255) & ~(size_t)255; }
constexpr size_t SZ_XE   = (size_t)MALL * DE * 4;
constexpr size_t SZ_XEB  = (size_t)MALL * DE * 2;
constexpr size_t SZ_WTC  = (size_t)DE * KCONV * 2;
constexpr size_t SZ_MEMB = (size_t)MEM_NP * DE * 2;
constexpr size_t SZ_T512 = (size_t)DE * DE * 2;
constexpr size_t SZ_W1T  = (size_t)CLSH * DCAT * 2;
constexpr size_t SZ_ATT  = (size_t)MHALF * MEM_NP * 2;
constexpr size_t SZ_TATT = (size_t)MHALF * 4;
constexpr size_t SZ_AUG  = (size_t)MHALF * DE * 2;
constexpr size_t SZ_F32H = (size_t)MHALF * DE * 4;
constexpr size_t SZ_XH   = (size_t)MALL * DE * 4;
constexpr size_t SZ_IDX  = AL((size_t)NBH * KSEL * 4);
constexpr size_t SZ_V16  = AL((size_t)16 * DE * 4);
constexpr size_t SZ_MAG  = AL((size_t)16 * KSEL * 4);
constexpr size_t SZ_POOL = (size_t)BN * DCAT * 4;
constexpr size_t SZ_CAH  = AL((size_t)BN * CAR * 4);
constexpr size_t SZ_SC   = (size_t)BN * T_ * 4;

constexpr size_t O_XE = 0;
constexpr size_t O_XEB    = O_XE + SZ_XE;
constexpr size_t O_WTC    = O_XEB + SZ_XEB;
constexpr size_t O_AMB    = O_WTC + SZ_WTC;
constexpr size_t O_NMB    = O_AMB + SZ_MEMB;
constexpr size_t O_AMT    = O_NMB + SZ_MEMB;
constexpr size_t O_NMT    = O_AMT + SZ_MEMB;
constexpr size_t O_MUT    = O_NMT + SZ_MEMB;
constexpr size_t O_VART   = O_MUT + SZ_T512;
constexpr size_t O_W1T    = O_VART + SZ_T512;
constexpr size_t O_ATT_AA = O_W1T + SZ_W1T;
constexpr size_t O_ATT_AN = O_ATT_AA + SZ_ATT;
constexpr size_t O_ATT_NA = O_ATT_AN + SZ_ATT;
constexpr size_t O_ATT_NN = O_ATT_NA + SZ_ATT;
constexpr size_t O_TAA    = O_ATT_NN + SZ_ATT;
constexpr size_t O_TAN    = O_TAA + SZ_TATT;
constexpr size_t O_TNA    = O_TAN + SZ_TATT;
constexpr size_t O_TNN    = O_TNA + SZ_TATT;
constexpr size_t O_AUG_AA = O_TNN + SZ_TATT;
constexpr size_t O_AUG_AN = O_AUG_AA + SZ_AUG;
constexpr size_t O_AUG_NA = O_AUG_AN + SZ_AUG;
constexpr size_t O_AUG_NN = O_AUG_NA + SZ_AUG;
constexpr size_t O_NNEW   = O_AUG_NN + SZ_AUG;
constexpr size_t O_ANEW   = O_NNEW + SZ_F32H;
constexpr size_t O_XH     = O_ANEW + SZ_F32H;
constexpr size_t O_IDXA   = O_XH + SZ_XH;
constexpr size_t O_IDXN   = O_IDXA + SZ_IDX;
constexpr size_t O_IDXP   = O_IDXN + SZ_IDX;
constexpr size_t O_VNEG   = O_IDXP + SZ_IDX;
constexpr size_t O_VANC   = O_VNEG + SZ_V16;
constexpr size_t O_VPOS   = O_VANC + SZ_V16;
constexpr size_t O_MAGN   = O_VPOS + SZ_V16;
constexpr size_t O_MAGA   = O_MAGN + SZ_MAG;
constexpr size_t O_POOL   = O_MAGA + SZ_MAG;
constexpr size_t O_CAH    = O_POOL + SZ_POOL;
constexpr size_t O_CATT   = O_CAH + SZ_CAH;
constexpr size_t O_SCE    = O_CATT + SZ_POOL;
constexpr size_t O_SCS    = O_SCE + SZ_SC;
constexpr size_t O_ACC    = O_SCS + SZ_SC;

// d_out layout (floats)
constexpr int OO_FRAME = 0, OO_TRIP = 8192, OO_KL = 8193, OO_DIST = 8194;
constexpr int OO_AATT = 8195, OO_NATT = OO_AATT + 4096, OO_ANATT = OO_NATT + 4096,
              OO_NAATT = OO_ANATT + 4096, OO_BEE = OO_NAATT + 4096,
              OO_BES = OO_BEE + 32, OO_BSE = OO_BES + 32, OO_BSS = OO_BSE + 32;

extern "C" void kernel_launch(void* const* d_in, const int* in_sizes, int n_in,
                              void* d_out, int out_size, void* d_ws, size_t ws_size,
                              hipStream_t stream) {
  const float* x      = (const float*)d_in[0];
  const float* conv_w = (const float*)d_in[1];
  const float* conv_b = (const float*)d_in[2];
  const float* a_mem  = (const float*)d_in[3];
  const float* n_mem  = (const float*)d_in[4];
  const float* mu_w   = (const float*)d_in[5];
  const float* mu_b   = (const float*)d_in[6];
  const float* var_w  = (const float*)d_in[7];
  const float* var_b  = (const float*)d_in[8];
  const float* eps    = (const float*)d_in[9];
  const float* ca_w1  = (const float*)d_in[10];
  const float* ca_b1  = (const float*)d_in[11];
  const float* ca_w2  = (const float*)d_in[12];
  const float* ca_b2  = (const float*)d_in[13];
  const float* cls_w1 = (const float*)d_in[14];
  const float* cls_b1 = (const float*)d_in[15];
  const float* cls_w2 = (const float*)d_in[16];
  const float* cls_b2 = (const float*)d_in[17];
  (void)in_sizes; (void)n_in; (void)out_size; (void)ws_size;

  char* ws = (char*)d_ws;
  float*  out   = (float*)d_out;
  float*  xe    = (float*)(ws + O_XE);
  bf16_t* xeb   = (bf16_t*)(ws + O_XEB);
  bf16_t* wtc   = (bf16_t*)(ws + O_WTC);
  bf16_t* amb   = (bf16_t*)(ws + O_AMB);
  bf16_t* nmb   = (bf16_t*)(ws + O_NMB);
  bf16_t* amT   = (bf16_t*)(ws + O_AMT);
  bf16_t* nmT   = (bf16_t*)(ws + O_NMT);
  bf16_t* muT   = (bf16_t*)(ws + O_MUT);
  bf16_t* varT  = (bf16_t*)(ws + O_VART);
  bf16_t* w1T   = (bf16_t*)(ws + O_W1T);
  bf16_t* attAA = (bf16_t*)(ws + O_ATT_AA);
  bf16_t* attAN = (bf16_t*)(ws + O_ATT_AN);
  bf16_t* attNA = (bf16_t*)(ws + O_ATT_NA);
  bf16_t* attNN = (bf16_t*)(ws + O_ATT_NN);
  float*  tAA   = (float*)(ws + O_TAA);
  float*  tAN   = (float*)(ws + O_TAN);
  float*  tNA   = (float*)(ws + O_TNA);
  float*  tNN   = (float*)(ws + O_TNN);
  bf16_t* augAA = (bf16_t*)(ws + O_AUG_AA);
  bf16_t* augAN = (bf16_t*)(ws + O_AUG_AN);
  bf16_t* augNA = (bf16_t*)(ws + O_AUG_NA);
  bf16_t* augNN = (bf16_t*)(ws + O_AUG_NN);
  float*  nnew  = (float*)(ws + O_NNEW);
  float*  anew  = (float*)(ws + O_ANEW);
  float*  xh    = (float*)(ws + O_XH);
  int*    idxA  = (int*)(ws + O_IDXA);
  int*    idxN  = (int*)(ws + O_IDXN);
  int*    idxP  = (int*)(ws + O_IDXP);
  float*  vneg  = (float*)(ws + O_VNEG);
  float*  vanc  = (float*)(ws + O_VANC);
  float*  vpos  = (float*)(ws + O_VPOS);
  float*  magN  = (float*)(ws + O_MAGN);
  float*  magA  = (float*)(ws + O_MAGA);
  float*  pool  = (float*)(ws + O_POOL);
  float*  cah   = (float*)(ws + O_CAH);
  float*  catt  = (float*)(ws + O_CATT);
  float*  sce   = (float*)(ws + O_SCE);
  float*  scs   = (float*)(ws + O_SCS);
  float*  acc   = (float*)(ws + O_ACC);

  const bf16_t* xeb_abn = xeb + (size_t)MHALF * DE;
  const float*  xe_abn  = xe + (size_t)MHALF * DE;

  auto blocks = [](size_t n, int b) { return (unsigned)((n + b - 1) / b); };

  // prep
  k_zero<<<1, 64, 0, stream>>>(acc, 16);
  k_prep_conv<<<blocks((size_t)DE * KCONV, 256), 256, 0, stream>>>(conv_w, wtc);
  k_prep_mem<<<blocks((size_t)MEM_NP * DE, 256), 256, 0, stream>>>(a_mem, n_mem, amb, nmb, amT, nmT);
  k_prep_t512<<<blocks((size_t)DE * DE, 256), 256, 0, stream>>>(mu_w, var_w, muT, varT);
  k_prep_w1<<<blocks((size_t)DCAT * CLSH, 256), 256, 0, stream>>>(cls_w1, w1T);

  // conv + relu (im2col GEMM, K=3072)
  k_conv_gemm<<<(MALL / 16) * (DE / 16) / 8, 256, 0, stream>>>(x, wtc, conv_b, xe, xeb);

  // attention GEMMs (4 combos)
  unsigned gatt = (MHALF / 16) * (MEM_NP / 16) / 8;
  k_att_gemm<<<gatt, 256, 0, stream>>>(xeb_abn, amb, attAA);
  k_att_gemm<<<gatt, 256, 0, stream>>>(xeb_abn, nmb, attAN);
  k_att_gemm<<<gatt, 256, 0, stream>>>(xeb,     amb, attNA);
  k_att_gemm<<<gatt, 256, 0, stream>>>(xeb,     nmb, attNN);

  // t_att (top4 mean)
  unsigned gt = blocks(MHALF, 256);
  k_tatt<<<gt, 256, 0, stream>>>(attAA, tAA);
  k_tatt<<<gt, 256, 0, stream>>>(attAN, tAN);
  k_tatt<<<gt, 256, 0, stream>>>(attNA, tNA);
  k_tatt<<<gt, 256, 0, stream>>>(attNN, tNN);

  // aug GEMMs
  unsigned gaug = (MHALF / 16) * (DE / 16) / 8;
  k_aug_gemm<<<gaug, 256, 0, stream>>>(attAA, amT, augAA);
  k_aug_gemm<<<gaug, 256, 0, stream>>>(attAN, nmT, augAN);
  k_aug_gemm<<<gaug, 256, 0, stream>>>(attNA, amT, augNA);
  k_aug_gemm<<<gaug, 256, 0, stream>>>(attNN, nmT, augNN);

  // VAE head + projections
  k_muvar<<<gaug, 256, 0, stream>>>(augNN, muT, varT, mu_b, var_b, eps, nnew, acc);
  k_gemm512<<<gaug, 256, 0, stream>>>(augAA, muT, mu_b, nullptr, anew);
  k_gemm512<<<gaug, 256, 0, stream>>>(augNA, muT, mu_b, nnew, xh);
  k_gemm512<<<gaug, 256, 0, stream>>>(augAN, muT, mu_b, anew, xh + (size_t)MHALF * DE);

  // top-17 + gathers + triplet / distance / kl
  k_topk<<<1, 128, 0, stream>>>(tAA, idxA);
  k_topk<<<1, 128, 0, stream>>>(tNN, idxN);
  k_topk<<<1, 128, 0, stream>>>(tAN, idxP);
  k_gathermean<<<blocks(16 * DE, 256), 256, 0, stream>>>(xe_abn, idxA, vneg);
  k_gathermean<<<blocks(16 * DE, 256), 256, 0, stream>>>(xe,     idxN, vanc);
  k_gathermean<<<blocks(16 * DE, 256), 256, 0, stream>>>(xe_abn, idxP, vpos);
  k_triplet<<<1, 256, 0, stream>>>(vanc, vpos, vneg, out + OO_TRIP);
  k_mag<<<16 * KSEL, 32, 0, stream>>>(nnew, idxN, magN);
  k_mag<<<16 * KSEL, 32, 0, stream>>>(anew, idxA, magA);
  k_distance<<<1, 256, 0, stream>>>(magN, magA, out + OO_DIST);
  k_klfin<<<1, 1, 0, stream>>>(acc, out + OO_KL);

  // channel attention + classifier
  k_pool<<<blocks((size_t)BN * DCAT, 256), 256, 0, stream>>>(xe, xh, pool);
  k_cahid<<<blocks((size_t)BN * CAR, 256), 256, 0, stream>>>(pool, ca_w1, ca_b1, cah);
  k_catt<<<blocks((size_t)BN * DCAT, 256), 256, 0, stream>>>(cah, ca_w2, ca_b2, catt);
  unsigned gcls = (MALL / 16) / 8;
  k_cls<<<gcls, 256, 0, stream>>>(xe, xh, catt, w1T, cls_b1, cls_w2, cls_b2, sce, 0);
  k_cls<<<gcls, 256, 0, stream>>>(xe, xh, catt, w1T, cls_b1, cls_w2, cls_b2, scs, 1);

  // outputs: attention means, frame, bags
  unsigned gam = blocks(16 * T_, 256);
  k_attmean<<<gam, 256, 0, stream>>>(tAA, out + OO_AATT);
  k_attmean<<<gam, 256, 0, stream>>>(tNN, out + OO_NATT);
  k_attmean<<<gam, 256, 0, stream>>>(tNA, out + OO_ANATT);
  k_attmean<<<gam, 256, 0, stream>>>(tAN, out + OO_NAATT);
  k_final<<<32, 256, 0, stream>>>(sce, scs, tNA, tAA, out + OO_FRAME,
                                  out + OO_BEE, out + OO_BES, out + OO_BSE, out + OO_BSS);
}
